// EdgeDecoderDotProduct_9844065042710
// MI455X (gfx1250) — matrix operand order repaired and checked by reading, not compiled, not run on hardware
//
#include <hip/hip_runtime.h>

typedef _Float16 v16h __attribute__((ext_vector_type(16)));
typedef float    v8f  __attribute__((ext_vector_type(8)));
typedef int      v4i  __attribute__((vector_size(16)));   // matches builtin's int4 pointee

constexpr int NB = 8;       // batch
constexpr int NN = 2048;    // nodes
constexpr int DM = 512;     // d_model
constexpr int DC = 128;     // coords dim
constexpr int FI = 640;     // fan_in
constexpr int LDA = 80;     // padded LDS row stride in halves (160B, 16B-aligned)

union Frag32 { v16h h; uint4 u[2]; };

#define AS1 __attribute__((address_space(1)))
#define AS3 __attribute__((address_space(3)))

#if defined(__AMDGCN__) && __has_builtin(__builtin_amdgcn_global_load_async_to_lds_b128)
#define HAVE_ASYNC_LDS 1
#else
#define HAVE_ASYNC_LDS 0
#endif

// 16B global -> LDS copy: async (ASYNCcnt) when available, register relay otherwise
__device__ __forceinline__ void cp_g2l_b128(_Float16* lds, const _Float16* g) {
#if HAVE_ASYNC_LDS
    __builtin_amdgcn_global_load_async_to_lds_b128((AS1 v4i*)g, (AS3 v4i*)lds, 0, 0);
#else
    *(uint4*)lds = *(const uint4*)g;
#endif
}

__device__ __forceinline__ void wait_async_copies() {
#if HAVE_ASYNC_LDS
#if __has_builtin(__builtin_amdgcn_s_wait_asynccnt)
    __builtin_amdgcn_s_wait_asynccnt(0);
#else
    asm volatile("s_wait_asynccnt 0" ::: "memory");
#endif
#endif
}

#define WMMA_F16(a, b, c) \
    __builtin_amdgcn_wmma_f32_16x16x32_f16(false, (a), false, (b), (short)0, (c), false, false)

// ---------------------------------------------------------------------------
// Fragment loaders (CDNA5 ISA 7.12.2, 16-bit A 16x32 / B 32x16). Work for both
// global and LDS pointers (addrspace inferred after inlining -> ds_load_b128).
// ---------------------------------------------------------------------------
__device__ __forceinline__ v16h load_a_f16(const _Float16* base, int ld, int k0, int lane) {
    int r  = lane & 15;
    int kb = k0 + ((lane >> 4) << 3);
    const _Float16* p = base + (size_t)r * ld + kb;
    Frag32 f;
    f.u[0] = *(const uint4*)(p);        // K kb..kb+7
    f.u[1] = *(const uint4*)(p + 16);   // K kb+16..kb+23
    return f.h;
}

__device__ __forceinline__ v16h load_b_f16(const _Float16* base, int ld, int k0, int lane) {
    int c  = lane & 15;
    int kb = k0 + ((lane >> 4) << 3);
    const _Float16* p = base + (size_t)c * ld + kb;
    Frag32 f;
    f.u[0] = *(const uint4*)(p);        // K kb..kb+7
    f.u[1] = *(const uint4*)(p + 16);   // K kb+16..kb+23
    return f.h;
}

__device__ __forceinline__ void cvt8(v16h& h, int o, float4 x, float4 y, float rs) {
    h[o+0] = (_Float16)(x.x*rs); h[o+1] = (_Float16)(x.y*rs);
    h[o+2] = (_Float16)(x.z*rs); h[o+3] = (_Float16)(x.w*rs);
    h[o+4] = (_Float16)(y.x*rs); h[o+5] = (_Float16)(y.y*rs);
    h[o+6] = (_Float16)(y.z*rs); h[o+7] = (_Float16)(y.w*rs);
}

__device__ __forceinline__ v16h load_a_cvt(const float* base, int ld, int k0, float rs, int lane) {
    int r  = lane & 15;
    int kb = k0 + ((lane >> 4) << 3);
    const float* p = base + (size_t)r * ld + kb;
    v16h h;
    cvt8(h, 0, *(const float4*)(p),      *(const float4*)(p + 4),  rs);
    cvt8(h, 8, *(const float4*)(p + 16), *(const float4*)(p + 20), rs);
    return h;
}

__device__ __forceinline__ v16h load_b_cvt(const float* base, int ld, int k0, float rs, int lane) {
    int c  = lane & 15;
    int kb = k0 + ((lane >> 4) << 3);
    const float* p = base + (size_t)c * ld + kb;
    v16h h;
    cvt8(h, 0, *(const float4*)(p),      *(const float4*)(p + 4),  rs);
    cvt8(h, 8, *(const float4*)(p + 16), *(const float4*)(p + 20), rs);
    return h;
}

// ---------------------------------------------------------------------------
// Kernel 1: x = (concat(embeds*mask, coords) @ W^T) * mask  -> f16 workspace
// ---------------------------------------------------------------------------
__global__ __launch_bounds__(256)
void linear_wmma_kernel(const float* __restrict__ embeds,
                        const float* __restrict__ coords,
                        const int*   __restrict__ mask,
                        const float* __restrict__ W,
                        _Float16*    __restrict__ xout) {
    const int wave = (int)((blockIdx.x * blockDim.x + threadIdx.x) >> 5);
    const int lane = threadIdx.x & 31;
    const int tilesPerB = (NN/16) * (DM/16);     // 4096
    const int b  = wave / tilesPerB;
    const int t  = wave % tilesPerB;
    const int tn = t / (DM/16);
    const int td = t % (DM/16);

    const int   arow0 = tn * 16;
    const float mrow  = (mask[b*NN + arow0 + (lane & 15)] != 0) ? 1.0f : 0.0f;

    const float* Ae = embeds + ((size_t)b*NN + arow0) * DM;
    const float* Ac = coords + ((size_t)b*NN + arow0) * DC;
    const float* Bw = W + (size_t)(td * 16) * FI;

    v8f acc = {};
    for (int k0 = 0; k0 < FI; k0 += 32) {        // 512 % 32 == 0: no concat straddle
        v16h a = (k0 < DM) ? load_a_cvt(Ae, DM, k0,      mrow, lane)
                           : load_a_cvt(Ac, DC, k0 - DM, 1.0f, lane);
        v16h w = load_b_cvt(Bw, FI, k0, 1.0f, lane);
        acc = WMMA_F16(a, w, acc);
    }

    const int hi  = lane >> 4;
    const int col = td * 16 + (lane & 15);
    _Float16* xb = xout + ((size_t)b*NN + arow0) * DM;
    #pragma unroll
    for (int j = 0; j < 8; ++j) {
        int   m  = j + 8*hi;
        float rm = (mask[b*NN + arow0 + m] != 0) ? 1.0f : 0.0f;
        xb[(size_t)m * DM + col] = (_Float16)(acc[j] * rm);
    }
}

// ---------------------------------------------------------------------------
// Kernel 2 epilogue helper: one 16x16 logits tile -> sigmoid/mask/triu outputs
// ---------------------------------------------------------------------------
__device__ __forceinline__ void emit_tile(const v8f& acc, int row0, int col0,
                                          const int* __restrict__ maskb,
                                          float s, float b0,
                                          float* __restrict__ pb,
                                          float* __restrict__ ab, int lane) {
    const int hi  = lane >> 4;
    const int col = col0 + (lane & 15);
    const int cm  = maskb[col];
    #pragma unroll
    for (int j = 0; j < 8; ++j) {
        int row = row0 + j + 8*hi;
        int rm  = maskb[row];
        float p = 0.0f;                              // sigmoid(-inf) for masked pairs
        if (rm && cm) {
            float logit = acc[j] * s + b0;
            p = 1.0f / (1.0f + __expf(-logit));
        }
        size_t idx = (size_t)row * NN + col;
        pb[idx] = p;
        ab[idx] = (p > 0.5f && col > row) ? 1.0f : 0.0f;
    }
}

// ---------------------------------------------------------------------------
// Kernel 2: logits = x @ x^T, LDS-blocked.
//  block = 8 waves (2x4) -> 64x128 supertile; wave -> 32x32 (2x2 WMMA tiles)
//  per K=64 step: stage A(64x64) + B(128x64) halves to LDS (async), 8 WMMAs/wave
// ---------------------------------------------------------------------------
__global__ __launch_bounds__(256)
void pairwise_wmma_kernel(const _Float16* __restrict__ x,
                          const int*   __restrict__ mask,
                          const float* __restrict__ scale,
                          const float* __restrict__ bias,
                          float* __restrict__ probs,
                          float* __restrict__ adj) {
    __shared__ __align__(16) _Float16 Abuf[64  * LDA];   // 10 KB
    __shared__ __align__(16) _Float16 Bbuf[128 * LDA];   // 20 KB

    const int tid  = threadIdx.x;
    const int lane = tid & 31;
    const int w    = tid >> 5;
    const int wr   = w >> 2;          // 0..1 : row wave
    const int wc   = w & 3;           // 0..3 : col wave

    const int tilesPerB = (NN/64) * (NN/128);            // 32*16 = 512
    const int b   = blockIdx.x / tilesPerB;
    const int t   = blockIdx.x % tilesPerB;
    const int rowBase = (t / (NN/128)) * 64;
    const int colBase = (t % (NN/128)) * 128;

    const _Float16* xb = x + (size_t)b * NN * DM;
    const _Float16* gA = xb + (size_t)rowBase * DM;
    const _Float16* gB = xb + (size_t)colBase * DM;

    v8f acc00 = {}, acc01 = {}, acc10 = {}, acc11 = {};

    for (int ks = 0; ks < DM; ks += 64) {
        // ---- stage A: 64 rows x 64 halves = 512 x 16B chunks, 2 per thread
        #pragma unroll
        for (int i = 0; i < 2; ++i) {
            int c = tid + i * 256;
            int row = c >> 3, ko = (c & 7) << 3;
            cp_g2l_b128(&Abuf[row * LDA + ko], gA + (size_t)row * DM + ks + ko);
        }
        // ---- stage B: 128 rows x 64 halves = 1024 x 16B chunks, 4 per thread
        #pragma unroll
        for (int i = 0; i < 4; ++i) {
            int c = tid + i * 256;
            int row = c >> 3, ko = (c & 7) << 3;
            cp_g2l_b128(&Bbuf[row * LDA + ko], gB + (size_t)row * DM + ks + ko);
        }
        wait_async_copies();
        __syncthreads();

        // ---- compute: 2 K-chunks x (2x2 tiles) = 8 WMMAs
        #pragma unroll
        for (int kk = 0; kk < 64; kk += 32) {
            v16h a0 = load_a_f16(&Abuf[(wr*32 +  0) * LDA], LDA, kk, lane);
            v16h a1 = load_a_f16(&Abuf[(wr*32 + 16) * LDA], LDA, kk, lane);
            v16h b0 = load_b_f16(&Bbuf[(wc*32 +  0) * LDA], LDA, kk, lane);
            v16h b1 = load_b_f16(&Bbuf[(wc*32 + 16) * LDA], LDA, kk, lane);
            acc00 = WMMA_F16(a0, b0, acc00);
            acc01 = WMMA_F16(a0, b1, acc01);
            acc10 = WMMA_F16(a1, b0, acc10);
            acc11 = WMMA_F16(a1, b1, acc11);
        }
        __syncthreads();
    }

    const float s  = scale[0];
    const float b0 = bias[0];
    const int*  maskb = mask + b * NN;
    float* pb = probs + (size_t)b * NN * NN;
    float* ab = adj   + (size_t)b * NN * NN;
    const int r0 = rowBase + wr * 32;
    const int c0 = colBase + wc * 32;
    emit_tile(acc00, r0,      c0,      maskb, s, b0, pb, ab, lane);
    emit_tile(acc01, r0,      c0 + 16, maskb, s, b0, pb, ab, lane);
    emit_tile(acc10, r0 + 16, c0,      maskb, s, b0, pb, ab, lane);
    emit_tile(acc11, r0 + 16, c0 + 16, maskb, s, b0, pb, ab, lane);
}

// ---------------------------------------------------------------------------
extern "C" void kernel_launch(void* const* d_in, const int* in_sizes, int n_in,
                              void* d_out, int out_size, void* d_ws, size_t ws_size,
                              hipStream_t stream) {
    (void)in_sizes; (void)n_in; (void)out_size; (void)ws_size;
    const float* embeds = (const float*)d_in[0];   // [8,2048,512] f32
    const float* coords = (const float*)d_in[1];   // [8,2048,128] f32
    const int*   mask   = (const int*)d_in[2];     // [8,2048,1]
    const float* W      = (const float*)d_in[3];   // [512,640]
    const float* scale  = (const float*)d_in[4];   // [1]
    const float* bias   = (const float*)d_in[5];   // [1]

    _Float16* x16   = (_Float16*)d_ws;              // 16 MB scratch
    float*    probs = (float*)d_out;                // [8,2048,2048]
    float*    adj   = probs + (size_t)NB * NN * NN; // [8,2048,2048] 0/1

    {
        const int waves = NB * (NN/16) * (DM/16);   // 32768 tiles, 1 wave each
        linear_wmma_kernel<<<waves * 32 / 256, 256, 0, stream>>>(
            embeds, coords, mask, W, x16);
    }
    {
        const int blocks = NB * (NN/64) * (NN/128); // 4096 supertiles
        pairwise_wmma_kernel<<<blocks, 256, 0, stream>>>(
            x16, mask, scale, bias, probs, adj);
    }
}
